// PAM_Module_893353198214
// MI455X (gfx1250) — compile-verified, matrix operand order checked by reading, not run on hardware
//
#include <hip/hip_runtime.h>
#include <hip/hip_bf16.h>
#include <math.h>

#define HW    4096
#define C_IN  256
#define C4_   64
#define BATCH 4
#define TM    128
#define TN    128

typedef __attribute__((ext_vector_type(16))) _Float16 v16h;
typedef __attribute__((ext_vector_type(8)))  _Float16 v8h;
typedef __attribute__((ext_vector_type(8)))  float    v8f;

// ---------- WMMA helpers -------------------------------------------------

__device__ __forceinline__ v8f wmma_f16(v16h a, v16h b, v8f c) {
  // D = A(16x32 f16) * B(32x16 f16) + C(16x16 f32)
  return __builtin_amdgcn_wmma_f32_16x16x32_f16(
      /*neg_a=*/false, a, /*neg_b=*/false, b,
      /*c_mod=*/(short)0, c, /*reuse_a=*/false, /*reuse_b=*/false);
}

// Load a 16x32 fragment from a row-major [row][k] tile (k contiguous),
// two 16-byte loads per lane. Works for LDS or global pointers.
// ISA layout: lanes 0-15 -> row = lane, k in {k0..k0+7, k0+16..k0+23};
//             lanes 16-31 -> row = lane-16, k in {k0+8..+15, k0+24..+31}.
__device__ __forceinline__ v16h frag_ld(const _Float16* t, int ld, int r0, int k0) {
  int lane = threadIdx.x & 31;
  int r = r0 + (lane & 15);
  int h = (lane >> 4) << 3;              // 0 or 8
  const _Float16* p = t + r * ld + k0 + h;
  v8h lo = *(const v8h*)(p);
  v8h hi = *(const v8h*)(p + 16);
  v16h f;
#pragma unroll
  for (int i = 0; i < 8; ++i) { f[i] = lo[i]; f[i + 8] = hi[i]; }
  return f;
}

__device__ __forceinline__ v8f vzero8() { v8f z;
#pragma unroll
  for (int i = 0; i < 8; ++i) z[i] = 0.0f; return z; }

// CDNA5 async global->LDS 16B copy (per-lane), tracked by ASYNCcnt.
__device__ __forceinline__ void async_b128(unsigned lds_off, const _Float16* g) {
  asm volatile("global_load_async_to_lds_b128 %0, %1, off"
               :: "v"(lds_off), "v"(g) : "memory");
}
__device__ __forceinline__ void wait_async8() {
  asm volatile("s_wait_asynccnt 0x8" ::: "memory");
}
__device__ __forceinline__ void wait_async0() {
  asm volatile("s_wait_asynccnt 0x0" ::: "memory");
}
__device__ __forceinline__ unsigned lds_off_of(const void* p) {
  // Flat LDS addresses carry the LDS byte offset in the low 32 bits.
  return (unsigned)(uintptr_t)p;
}

// monotonic float<->uint for atomicMax over signed floats
__device__ __forceinline__ unsigned f2key(float f) {
  unsigned u = __float_as_uint(f);
  return (u & 0x80000000u) ? ~u : (u | 0x80000000u);
}
__device__ __forceinline__ float key2f(unsigned k) {
  return (k & 0x80000000u) ? __uint_as_float(k & 0x7FFFFFFFu)
                           : __uint_as_float(~k);
}

// ---------- kernels ------------------------------------------------------

__global__ void init_kernel(unsigned* Mkey, float* Z) {
  int t = threadIdx.x;
  if (t < BATCH) { Mkey[t] = 0u; Z[t] = 0.0f; }
}

// one-time fp32 -> f16 conversion of the three projection weights
__global__ void __launch_bounds__(256) wconv_kernel(
    const float* __restrict__ Wq, const float* __restrict__ Wk,
    const float* __restrict__ Wv, _Float16* __restrict__ Wh) {
  int i = blockIdx.x * 256 + threadIdx.x;
  if (i < C4_ * C_IN) {
    Wh[i]                   = (_Float16)Wq[i];
    Wh[C4_ * C_IN + i]      = (_Float16)Wk[i];
    Wh[2 * C4_ * C_IN + i]  = (_Float16)Wv[i];
  }
}

// Q/K/V projection via WMMA. Q,K stored [b][hw][c4] (c contiguous, packed
// 16B f16 stores); V stored [b][c4][hw] for the PV A-fragments.
__global__ void __launch_bounds__(256) qkv_kernel(
    const float* __restrict__ x, const _Float16* __restrict__ Wh,
    const float* __restrict__ bq, const float* __restrict__ bk,
    const float* __restrict__ bv,
    _Float16* __restrict__ Qh, _Float16* __restrict__ Kh,
    _Float16* __restrict__ Vh) {
  __shared__ __align__(16) _Float16 Xt[TN][C_IN];   // 64 KB, x tile transposed
  int b  = blockIdx.y;
  int n0 = blockIdx.x * TN;
  int t  = threadIdx.x;
  for (int idx = t; idx < TN * C_IN; idx += 256) {
    int c = idx / TN, n = idx % TN;                 // consecutive t -> consec. n
    Xt[n][c] = (_Float16)x[((size_t)b * C_IN + c) * HW + n0 + n];
  }
  __syncthreads();
  int w = t >> 5, lane = t & 31;
  int cs = (w & 3) * 16;        // output-channel strip (M)
  int nh = (w >> 2) * 64;       // n half (4 n-tiles of 16)
  const float* bs[3] = {bq, bk, bv};
#pragma unroll
  for (int p = 0; p < 3; ++p) {
    const _Float16* Wp = Wh + p * C4_ * C_IN;
    v8f acc[4];
#pragma unroll
    for (int j = 0; j < 4; ++j) acc[j] = vzero8();
#pragma unroll
    for (int kk = 0; kk < C_IN; kk += 32) {
      v16h a = frag_ld(Wp, C_IN, cs, kk);           // f16 weights, 2x16B loads
#pragma unroll
      for (int j = 0; j < 4; ++j) {
        v16h bf = frag_ld(&Xt[0][0], C_IN, nh + j * 16, kk);
        acc[j] = wmma_f16(a, bf, acc[j]);
      }
    }
    int c0 = cs + 8 * (lane >> 4);                  // C-frag: M = r + 8*(lane>=16)
    float bb[8];
#pragma unroll
    for (int r = 0; r < 8; ++r) bb[r] = bs[p][c0 + r];
#pragma unroll
    for (int j = 0; j < 4; ++j) {
      int n = n0 + nh + j * 16 + (lane & 15);
      v8h ph;
#pragma unroll
      for (int r = 0; r < 8; ++r) ph[r] = (_Float16)(acc[j][r] + bb[r]);
      if (p < 2) {            // Q,K: [b][n][c] layout -> one 16B store
        _Float16* dst = (p == 0 ? Qh : Kh) + ((size_t)b * HW + n) * C4_ + c0;
        *(v8h*)dst = ph;
      } else {                // V: [b][c][n] layout
#pragma unroll
        for (int r = 0; r < 8; ++r)
          Vh[((size_t)b * C4_ + c0 + r) * HW + n] = ph[r];
      }
    }
  }
}

// Pass 1: S = Q^T K tiles (128x128) via WMMA, global max per batch.
// Tiles pulled in with async global->LDS b128 copies.
__global__ void __launch_bounds__(256) smax_kernel(
    const _Float16* __restrict__ Qh, const _Float16* __restrict__ Kh,
    unsigned* __restrict__ Mkey) {
  __shared__ __align__(16) _Float16 Qt[TM][C4_];   // [m][c] 16 KB
  __shared__ __align__(16) _Float16 Kt[TN][C4_];   // [n][c] 16 KB
  __shared__ float red[8];
  int b = blockIdx.z;
  int m0 = blockIdx.y * TM, n0 = blockIdx.x * TN;
  int t = threadIdx.x, w = t >> 5;
  const _Float16* Qb = Qh + (size_t)b * HW * C4_;  // [hw][c]
  const _Float16* Kb = Kh + (size_t)b * HW * C4_;
  unsigned qtl = lds_off_of(&Qt[0][0]);
  unsigned ktl = lds_off_of(&Kt[0][0]);
#pragma unroll
  for (int i = 0; i < 4; ++i) {                    // 1024 16B chunks per tile
    int id = t + i * 256;
    int r = id >> 3, c16 = (id & 7) * 8;
    async_b128(qtl + (unsigned)(r * C4_ + c16) * 2, Qb + (size_t)(m0 + r) * C4_ + c16);
    async_b128(ktl + (unsigned)(r * C4_ + c16) * 2, Kb + (size_t)(n0 + r) * C4_ + c16);
  }
  wait_async0();
  __syncthreads();
  v8f acc[8];
#pragma unroll
  for (int j = 0; j < 8; ++j) acc[j] = vzero8();
#pragma unroll
  for (int kk = 0; kk < C4_; kk += 32) {
    v16h a = frag_ld(&Qt[0][0], C4_, w * 16, kk);
#pragma unroll
    for (int j = 0; j < 8; ++j) {
      v16h bf = frag_ld(&Kt[0][0], C4_, j * 16, kk);
      acc[j] = wmma_f16(a, bf, acc[j]);
    }
  }
  float mx = -3.0e38f;
#pragma unroll
  for (int j = 0; j < 8; ++j)
#pragma unroll
    for (int r = 0; r < 8; ++r) mx = fmaxf(mx, acc[j][r]);
#pragma unroll
  for (int off = 16; off > 0; off >>= 1)
    mx = fmaxf(mx, __shfl_xor(mx, off, 32));
  if ((t & 31) == 0) red[w] = mx;
  __syncthreads();
  if (t == 0) {
    float m = red[0];
#pragma unroll
    for (int i = 1; i < 8; ++i) m = fmaxf(m, red[i]);
    atomicMax(&Mkey[b], f2key(m));
  }
}

// Pass 2: recompute S tiles, P = exp(S - M), accumulate Z and O = V * P.
// Double-buffered async global->LDS prefetch of the Q/V m-tiles.
__global__ void __launch_bounds__(256) attn_kernel(
    const _Float16* __restrict__ Qh, const _Float16* __restrict__ Kh,
    const _Float16* __restrict__ Vh, const unsigned* __restrict__ Mkey,
    float* __restrict__ Z, float* __restrict__ Ounn) {
  __shared__ __align__(16) _Float16 Kt[TN][C4_];        // 16 KB (persistent)
  __shared__ __align__(16) _Float16 Qt[2][TM][C4_];     // 32 KB double buffer
  __shared__ __align__(16) _Float16 Vt[2][C4_][TM];     // 32 KB double buffer
  __shared__ __align__(16) _Float16 Pt[TN][TM];         // 32 KB (f16 probs)
  __shared__ float red[8];
  int b = blockIdx.y, n0 = blockIdx.x * TN;
  int t = threadIdx.x, w = t >> 5, lane = t & 31;
  const _Float16* Qb = Qh + (size_t)b * HW * C4_;       // [hw][c]
  const _Float16* Kb = Kh + (size_t)b * HW * C4_;       // [hw][c]
  const _Float16* Vb = Vh + (size_t)b * C4_ * HW;       // [c][hw]
  float Mb = key2f(Mkey[b]);
  unsigned ktl  = lds_off_of(&Kt[0][0]);
  unsigned qtl[2] = {lds_off_of(&Qt[0][0][0]), lds_off_of(&Qt[1][0][0])};
  unsigned vtl[2] = {lds_off_of(&Vt[0][0][0]), lds_off_of(&Vt[1][0][0])};

  // Kt once (4 async ops/thread), then tile 0 (8 async ops/thread)
#pragma unroll
  for (int i = 0; i < 4; ++i) {
    int id = t + i * 256;
    int n = id >> 3, c16 = (id & 7) * 8;
    async_b128(ktl + (unsigned)(n * C4_ + c16) * 2, Kb + (size_t)(n0 + n) * C4_ + c16);
  }
  auto prefetch_tile = [&](int m0, int buf) {
#pragma unroll
    for (int i = 0; i < 4; ++i) {                       // Qt: 128 rows x 8 chunks
      int id = t + i * 256;
      int m = id >> 3, c16 = (id & 7) * 8;
      async_b128(qtl[buf] + (unsigned)(m * C4_ + c16) * 2,
                 Qb + (size_t)(m0 + m) * C4_ + c16);
    }
#pragma unroll
    for (int i = 0; i < 4; ++i) {                       // Vt: 64 rows x 16 chunks
      int id = t + i * 256;
      int c = id >> 4, mh = (id & 15) * 8;
      async_b128(vtl[buf] + (unsigned)(c * TM + mh) * 2,
                 Vb + (size_t)c * HW + m0 + mh);
    }
  };
  prefetch_tile(0, 0);

  float zloc = 0.0f;
  int cs = (w & 3) * 16;     // V channel strip for PV phase
  int nh = (w >> 2) * 64;    // n half for PV phase
  v8f Oacc[4];
#pragma unroll
  for (int j = 0; j < 4; ++j) Oacc[j] = vzero8();

  const int NT = HW / TM;                               // 32 m-tiles
  for (int it = 0; it < NT; ++it) {
    int cur = it & 1;
    if (it + 1 < NT) {
      prefetch_tile((it + 1) * TM, cur ^ 1);            // overlap with compute
      wait_async8();                                    // tile `it` complete
    } else {
      wait_async0();
    }
    __syncthreads();
    // --- S tile: wave w owns m-strip w*16, all 128 n ---
    v8f acc[8];
#pragma unroll
    for (int j = 0; j < 8; ++j) acc[j] = vzero8();
#pragma unroll
    for (int kk = 0; kk < C4_; kk += 32) {
      v16h a = frag_ld(&Qt[cur][0][0], C4_, w * 16, kk);
#pragma unroll
      for (int j = 0; j < 8; ++j) {
        v16h bf = frag_ld(&Kt[0][0], C4_, j * 16, kk);
        acc[j] = wmma_f16(a, bf, acc[j]);
      }
    }
    // --- exp, local Z, store P transposed [n][m] as f16 ---
    int mloc = w * 16 + 8 * (lane >> 4);                // C-frag: M = r + 8*(lane>=16)
#pragma unroll
    for (int j = 0; j < 8; ++j) {
      v8h ph;
#pragma unroll
      for (int r = 0; r < 8; ++r) {
        float p = __expf(acc[j][r] - Mb);
        zloc += p;
        ph[r] = (_Float16)p;
      }
      *(v8h*)&Pt[j * 16 + (lane & 15)][mloc] = ph;      // 16B DS store
    }
    __syncthreads();
    // --- PV: O[c,n] += V[c,m] * P[m,n] ---
#pragma unroll
    for (int ks = 0; ks < TM; ks += 32) {
      v16h a = frag_ld(&Vt[cur][0][0], TM, cs, ks);
#pragma unroll
      for (int j = 0; j < 4; ++j) {
        v16h bf = frag_ld(&Pt[0][0], TM, nh + j * 16, ks);
        Oacc[j] = wmma_f16(a, bf, Oacc[j]);
      }
    }
    __syncthreads();   // PV done before next iteration's prefetch overwrites
  }
  // write unnormalized O
  float* Ob = Ounn + (size_t)b * C4_ * HW;
  int c0 = cs + 8 * (lane >> 4);
#pragma unroll
  for (int j = 0; j < 4; ++j) {
    int n = n0 + nh + j * 16 + (lane & 15);
#pragma unroll
    for (int r = 0; r < 8; ++r) Ob[(size_t)(c0 + r) * HW + n] = Oacc[j][r];
  }
  // global Z accumulation
#pragma unroll
  for (int off = 16; off > 0; off >>= 1) zloc += __shfl_xor(zloc, off, 32);
  if ((t & 31) == 0) red[w] = zloc;
  __syncthreads();
  if (t == 0) {
    float s = 0.0f;
#pragma unroll
    for (int i = 0; i < 8; ++i) s += red[i];
    atomicAdd(&Z[b], s);
  }
}

// Output conv (256x64, fp32) with 1/Z folded in, + gamma*out + x residual.
__global__ void __launch_bounds__(256) out_kernel(
    const float* __restrict__ x, const float* __restrict__ Wo,
    const float* __restrict__ bo, const float* __restrict__ gamma,
    const float* __restrict__ Ounn, const float* __restrict__ Z,
    float* __restrict__ out) {
  int b = blockIdx.z, co = blockIdx.y;
  int n = blockIdx.x * 256 + threadIdx.x;
  float invZ = 1.0f / Z[b];
  const float* Ob = Ounn + (size_t)b * C4_ * HW;
  const float* Wr = Wo + co * C4_;
  float s = 0.0f;
#pragma unroll 16
  for (int c = 0; c < C4_; ++c) s = fmaf(Wr[c], Ob[(size_t)c * HW + n], s);
  float o = s * invZ + bo[co];
  size_t xi = ((size_t)b * C_IN + co) * HW + n;
  out[xi] = gamma[0] * o + x[xi];
}

// ---------- launch -------------------------------------------------------

extern "C" void kernel_launch(void* const* d_in, const int* in_sizes, int n_in,
                              void* d_out, int out_size, void* d_ws, size_t ws_size,
                              hipStream_t stream) {
  const float* x     = (const float*)d_in[0];
  const float* Wq    = (const float*)d_in[1];
  const float* bq    = (const float*)d_in[2];
  const float* Wk    = (const float*)d_in[3];
  const float* bk    = (const float*)d_in[4];
  const float* Wv    = (const float*)d_in[5];
  const float* bv    = (const float*)d_in[6];
  const float* Wo    = (const float*)d_in[7];
  const float* bo    = (const float*)d_in[8];
  const float* gamma = (const float*)d_in[9];
  float* out = (float*)d_out;

  // workspace layout
  char* ws = (char*)d_ws;
  const size_t qkv_elems = (size_t)BATCH * C4_ * HW;      // 1 Mi elems
  _Float16* Qh   = (_Float16*)(ws);                       // [b][hw][c4] f16
  _Float16* Kh   = (_Float16*)(ws + qkv_elems * 2);       // [b][hw][c4] f16
  _Float16* Vh   = (_Float16*)(ws + qkv_elems * 4);       // [b][c4][hw] f16
  float*    Ounn = (float*)   (ws + qkv_elems * 6);       // [b][c4][hw] f32
  unsigned* Mkey = (unsigned*)(ws + qkv_elems * 6 + qkv_elems * 4);
  float*    Zp   = (float*)   ((char*)Mkey + 256);
  _Float16* Wh   = (_Float16*)((char*)Mkey + 512);        // 3x(64x256) f16

  init_kernel<<<1, 64, 0, stream>>>(Mkey, Zp);
  wconv_kernel<<<(C4_ * C_IN + 255) / 256, 256, 0, stream>>>(Wq, Wk, Wv, Wh);
  qkv_kernel<<<dim3(HW / TN, BATCH), 256, 0, stream>>>(
      x, Wh, bq, bk, bv, Qh, Kh, Vh);
  smax_kernel<<<dim3(HW / TN, HW / TM, BATCH), 256, 0, stream>>>(Qh, Kh, Mkey);
  attn_kernel<<<dim3(HW / TN, BATCH), 256, 0, stream>>>(Qh, Kh, Vh, Mkey, Zp, Ounn);
  out_kernel<<<dim3(HW / 256, C_IN, BATCH), 256, 0, stream>>>(
      x, Wo, bo, gamma, Ounn, Zp, out);
}
